// SelfAttention_48129403519271
// MI455X (gfx1250) — compile-verified, matrix operand order checked by reading, not run on hardware
//
#include <hip/hip_runtime.h>

// Self-attention: B=4, H=8, L=2048, D=128.  FP16 WMMA (f32 accum) throughout.
// Stage 1: f32->f16 convert; weights transposed so all WMMA fragment loads
//          are contiguous 16B vector loads.
// Stage 2: QKV projection GEMM -> q/k in [B,H,L,D], v transposed [B,H,D,L].
// Stage 3: flash attention, 64-key tiles, one wave per 16-query tile,
//          base-2 online softmax via raw v_exp_f32 (log2e folded into Q);
//          row-sums of P computed on the matrix pipe (P x ones).
// Stage 4: output projection GEMM + bias -> f32 out.

#define D_HEAD 128
#define NHEAD  8
#define BATCH  4
#define SEQ    2048
#define NROWS  (BATCH * SEQ)      // 8192
#define DMODEL (D_HEAD * NHEAD)   // 1024

typedef __attribute__((ext_vector_type(16))) _Float16 v16h;
typedef __attribute__((ext_vector_type(8)))  _Float16 v8h;
typedef __attribute__((ext_vector_type(8)))  float    v8f;

static __device__ __forceinline__ v8f wmma16(v16h a, v16h b, v8f c) {
  // (neg_a, A, neg_b, B, c_mod, C, reuse_a, reuse_b)
  return __builtin_amdgcn_wmma_f32_16x16x32_f16(false, a, false, b, (short)0, c,
                                                false, false);
}

// 16x32 fragment from a row-major f16 matrix: rows row0..row0+15 across lanes
// (lane&15), per-lane halves at k = k0 + 8*(lane>>4) + {0..7, 16..23}.
// This is simultaneously:
//   - an A-fragment (MxK) of the matrix, and
//   - a B-fragment (KxN) of its transpose.
// Requires 16B alignment: p 16B-aligned, ld multiple of 8, k0 multiple of 32.
static __device__ __forceinline__ v16h
frag_load(const _Float16* __restrict__ p, int ld, int row0, int k0, int lane) {
  const _Float16* base =
      p + (size_t)(row0 + (lane & 15)) * ld + k0 + ((lane >> 4) << 3);
  v8h lo = *(const v8h*)(base);        // k run +0..7   -> global_load_b128
  v8h hi = *(const v8h*)(base + 16);   // k run +16..23 -> global_load_b128
  return __builtin_shufflevector(lo, hi, 0, 1, 2, 3, 4, 5, 6, 7, 8, 9, 10, 11,
                                 12, 13, 14, 15);
}

// ---------------------------------------------------------------- stage 1
__global__ void cvt_f32_to_f16_kernel(const float* __restrict__ in,
                                      _Float16* __restrict__ out, int n) {
  int i = blockIdx.x * blockDim.x + threadIdx.x;
  int stride = gridDim.x * blockDim.x;
  for (; i < n; i += stride) out[i] = (_Float16)in[i];
}

// in[R][C] (f32) -> out[C][R] (f16)
__global__ void cvt_transpose_kernel(const float* __restrict__ in,
                                     _Float16* __restrict__ out, int R, int C) {
  int i = blockIdx.x * blockDim.x + threadIdx.x;
  int total = R * C;
  int stride = gridDim.x * blockDim.x;
  for (; i < total; i += stride) {
    int r = i / C, c = i - r * C;
    out[(size_t)c * R + r] = (_Float16)in[i];
  }
}

// ---------------------------------------------------------------- stage 2
// grid: (NROWS/16, NHEAD, 3), block 256 (8 waves); wave w owns 16 cols of one
// head.  K = 128 -> 4 WMMA steps.  Weights are pre-transposed [N][K] so the
// B-fragment is a contiguous frag_load.  q/k written [B,H,L,D]; v written
// transposed [B,H,D,L] so attention's P*V B-frags are contiguous.
__global__ __launch_bounds__(256) void qkv_proj_kernel(
    const _Float16* __restrict__ xh, const _Float16* __restrict__ wqT,
    const _Float16* __restrict__ wkT, const _Float16* __restrict__ wvT,
    _Float16* __restrict__ qb, _Float16* __restrict__ kb,
    _Float16* __restrict__ vbT) {
  const int lane = threadIdx.x & 31;
  const int wave = threadIdx.x >> 5;
  const int m0 = blockIdx.x * 16;
  const int h = blockIdx.y;
  const int which = blockIdx.z;
  const _Float16* wT = (which == 0) ? wqT : (which == 1) ? wkT : wvT;
  const int n0 = h * D_HEAD + wave * 16;   // column in [0, DMODEL)

  v8f c = {};
#pragma unroll
  for (int kk = 0; kk < 4; ++kk) {
    v16h a = frag_load(xh, D_HEAD, m0, kk * 32, lane);      // x[m][k]
    v16h b = frag_load(wT, D_HEAD, n0, kk * 32, lane);      // W^T[n][k]
    c = wmma16(a, b, c);
  }

  const int n = lane & 15;
  const int half = lane >> 4;
#pragma unroll
  for (int r = 0; r < 8; ++r) {
    int row = m0 + r + half * 8;       // row in [0, NROWS)
    int bidx = row >> 11;              // / SEQ
    int l = row & (SEQ - 1);
    int d = wave * 16 + n;
    size_t bh = (size_t)bidx * NHEAD + h;
    if (which == 2) {
      vbT[(bh * D_HEAD + d) * SEQ + l] = (_Float16)c[r];    // [B,H,D,L]
    } else {
      _Float16* ob = (which == 0) ? qb : kb;
      ob[(bh * SEQ + l) * D_HEAD + d] = (_Float16)c[r];     // [B,H,L,D]
    }
  }
}

// ---------------------------------------------------------------- stage 3
// grid: (SEQ/128, B*H), block 256 (8 waves); wave owns a 16-query tile and
// marches over keys in 64-wide tiles.  Scores carry log2(e)/sqrt(D) so the
// softmax runs in base-2 with bare v_exp_f32.  Row maxes use cross-lane
// shuffles; row sums ride the matrix pipe (P x ones).
__global__ __launch_bounds__(256) void flash_attn_kernel(
    const _Float16* __restrict__ qb, const _Float16* __restrict__ kb,
    const _Float16* __restrict__ vbT, _Float16* __restrict__ attn) {
  __shared__ alignas(16) _Float16 lds_p[8][16 * 64];  // per-wave P staging
  const int lane = threadIdx.x & 31;
  const int wave = threadIdx.x >> 5;
  const int bh = blockIdx.y;
  const int b = bh >> 3, h = bh & 7;
  const int q0 = blockIdx.x * 128 + wave * 16;

  const _Float16* Q  = qb  + (size_t)bh * SEQ * D_HEAD;   // [L][D]
  const _Float16* K  = kb  + (size_t)bh * SEQ * D_HEAD;   // [L][D]
  const _Float16* Vt = vbT + (size_t)bh * D_HEAD * SEQ;   // [D][L]

  // Fold log2(e)/sqrt(D) into Q: scores come out in log2 units.
  const _Float16 hscale =
      (_Float16)(0.08838834764831845f * 1.4426950408889634f);
  v16h qf[4];
#pragma unroll
  for (int kk = 0; kk < 4; ++kk) {
    qf[kk] = frag_load(Q, D_HEAD, q0, kk * 32, lane);
#pragma unroll
    for (int i = 0; i < 16; ++i) qf[kk][i] = qf[kk][i] * hscale;
  }

  v16h ones;
#pragma unroll
  for (int i = 0; i < 16; ++i) ones[i] = (_Float16)1.0f;

  v8f o[8];
#pragma unroll
  for (int t = 0; t < 8; ++t) o[t] = (v8f){};
  float mrun[8], lrun[8];
#pragma unroll
  for (int r = 0; r < 8; ++r) { mrun[r] = -1e30f; lrun[r] = 0.0f; }

  const int n = lane & 15, half = lane >> 4;

  for (int j = 0; j < SEQ; j += 64) {
    if (j + 64 < SEQ) {  // pull next K/V tiles toward cache
      __builtin_prefetch(K + (size_t)(j + 64 + lane) * D_HEAD, 0, 1);
      __builtin_prefetch(K + (size_t)(j + 96 + lane) * D_HEAD, 0, 1);
      __builtin_prefetch(Vt + (size_t)(lane) * SEQ + j + 64, 0, 1);
      __builtin_prefetch(Vt + (size_t)(lane + 64) * SEQ + j + 64, 0, 1);
    }

    // S2 = (Q*log2e/sqrtD) * K^T over 64 keys: 4 sub-tiles, K=128 -> 16 WMMAs.
    v8f s[4];
#pragma unroll
    for (int t = 0; t < 4; ++t) s[t] = (v8f){};
#pragma unroll
    for (int kk = 0; kk < 4; ++kk)
#pragma unroll
      for (int t = 0; t < 4; ++t)
        s[t] = wmma16(qf[kk], frag_load(K, D_HEAD, j + t * 16, kk * 32, lane),
                      s[t]);

    // Base-2 online softmax max + raw v_exp_f32; results go to LDS staging.
    float alpha[8];
#pragma unroll
    for (int r = 0; r < 8; ++r) {
      float tmax = fmaxf(fmaxf(s[0][r], s[1][r]), fmaxf(s[2][r], s[3][r]));
#pragma unroll
      for (int msk = 1; msk < 16; msk <<= 1)
        tmax = fmaxf(tmax, __shfl_xor(tmax, msk, 32));  // row spans 16 lanes
      float mn = fmaxf(mrun[r], tmax);
      alpha[r] = __builtin_amdgcn_exp2f(mrun[r] - mn);  // bare v_exp_f32
      mrun[r] = mn;
      int m = r + half * 8;
#pragma unroll
      for (int t = 0; t < 4; ++t)
        lds_p[wave][m * 64 + t * 16 + n] =
            (_Float16)__builtin_amdgcn_exp2f(s[t][r] - mn);
    }

#pragma unroll
    for (int t = 0; t < 8; ++t)
#pragma unroll
      for (int r = 0; r < 8; ++r) o[t][r] *= alpha[r];

    // Reload P (16x64) as two A-fragments (C-layout -> A-layout transpose).
    asm volatile("s_wait_dscnt 0x0" ::: "memory");  // wave-local LDS RAW
    const _Float16* pb = &lds_p[wave][(lane & 15) * 64 + ((lane >> 4) << 3)];
    v8h l0 = *(const v8h*)(pb);         // ds_load_b128 x4
    v8h h0 = *(const v8h*)(pb + 16);
    v8h l1 = *(const v8h*)(pb + 32);
    v8h h1 = *(const v8h*)(pb + 48);
    v16h pf0 = __builtin_shufflevector(l0, h0, 0, 1, 2, 3, 4, 5, 6, 7, 8, 9,
                                       10, 11, 12, 13, 14, 15);
    v16h pf1 = __builtin_shufflevector(l1, h1, 0, 1, 2, 3, 4, 5, 6, 7, 8, 9,
                                       10, 11, 12, 13, 14, 15);

    // Row sums on the matrix pipe: rsum[r] = sum_k P[m][k] (C-layout == lrun).
    v8f rsum = {};
    rsum = wmma16(pf0, ones, rsum);
    rsum = wmma16(pf1, ones, rsum);
#pragma unroll
    for (int r = 0; r < 8; ++r) lrun[r] = lrun[r] * alpha[r] + rsum[r];

    // O += P * V.  B-frag of V tile (32x16) == A-layout frag of V^T rows.
#pragma unroll
    for (int t = 0; t < 8; ++t) {
      o[t] = wmma16(pf0, frag_load(Vt, SEQ, t * 16, j, lane), o[t]);
      o[t] = wmma16(pf1, frag_load(Vt, SEQ, t * 16, j + 32, lane), o[t]);
    }
  }

  // epilogue: normalize (v_rcp_f32) and store to attn[B*L][DMODEL] f16
  float inv[8];
#pragma unroll
  for (int r = 0; r < 8; ++r) inv[r] = __builtin_amdgcn_rcpf(lrun[r]);
#pragma unroll
  for (int t = 0; t < 8; ++t) {
#pragma unroll
    for (int r = 0; r < 8; ++r) {
      int m = r + half * 8;
      size_t row = (size_t)b * SEQ + q0 + m;
      int col = h * D_HEAD + t * 16 + n;
      attn[row * DMODEL + col] = (_Float16)(o[t][r] * inv[r]);
    }
  }
}

// ---------------------------------------------------------------- stage 4
// grid: (NROWS/16), block 256; wave w owns cols w*16..w*16+15.  K=1024.
// Wu pre-transposed [D_HEAD][DMODEL] so B-frags are contiguous.
__global__ __launch_bounds__(256) void out_proj_kernel(
    const _Float16* __restrict__ attn, const _Float16* __restrict__ wuT,
    const float* __restrict__ bu, float* __restrict__ out) {
  const int lane = threadIdx.x & 31;
  const int wave = threadIdx.x >> 5;
  const int m0 = blockIdx.x * 16;
  const int n0 = wave * 16;

  v8f c = {};
#pragma unroll 4
  for (int k0 = 0; k0 < DMODEL; k0 += 32) {
    v16h a = frag_load(attn, DMODEL, m0, k0, lane);   // attn[m][k]
    v16h b = frag_load(wuT, DMODEL, n0, k0, lane);    // Wu^T[n][k]
    c = wmma16(a, b, c);
  }

  const int n = lane & 15, half = lane >> 4;
  float bias = bu[n0 + n];
#pragma unroll
  for (int r = 0; r < 8; ++r) {
    int row = m0 + r + half * 8;
    out[(size_t)row * D_HEAD + n0 + n] = c[r] + bias;
  }
}

// ---------------------------------------------------------------- launch
extern "C" void kernel_launch(void* const* d_in, const int* in_sizes, int n_in,
                              void* d_out, int out_size, void* d_ws,
                              size_t ws_size, hipStream_t stream) {
  (void)in_sizes; (void)n_in; (void)out_size; (void)ws_size;
  const float* x  = (const float*)d_in[0];
  const float* Wq = (const float*)d_in[1];
  const float* Wk = (const float*)d_in[2];
  const float* Wv = (const float*)d_in[3];
  const float* Wu = (const float*)d_in[4];
  const float* bu = (const float*)d_in[5];
  float* out = (float*)d_out;

  char* ws = (char*)d_ws;
  const size_t MB = 1u << 20;
  _Float16* qf   = (_Float16*)(ws + 0 * MB);    // 16 MB  [B,H,L,D]
  _Float16* kf   = (_Float16*)(ws + 16 * MB);   // 16 MB  [B,H,L,D]
  _Float16* vfT  = (_Float16*)(ws + 32 * MB);   // 16 MB  [B,H,D,L]
  _Float16* attn = (_Float16*)(ws + 48 * MB);   // 16 MB  [B*L, DMODEL]
  _Float16* xh   = (_Float16*)(ws + 64 * MB);   //  2 MB
  _Float16* wqT  = (_Float16*)(ws + 66 * MB);                 // [DMODEL][D]
  _Float16* wkT  = (_Float16*)(ws + 66 * MB + 256 * 1024);    // [DMODEL][D]
  _Float16* wvT  = (_Float16*)(ws + 66 * MB + 512 * 1024);    // [DMODEL][D]
  _Float16* wuT  = (_Float16*)(ws + 66 * MB + 768 * 1024);    // [D][DMODEL]

  cvt_f32_to_f16_kernel<<<1024, 256, 0, stream>>>(x, xh, NROWS * D_HEAD);
  cvt_transpose_kernel<<<256, 256, 0, stream>>>(Wq, wqT, D_HEAD, DMODEL);
  cvt_transpose_kernel<<<256, 256, 0, stream>>>(Wk, wkT, D_HEAD, DMODEL);
  cvt_transpose_kernel<<<256, 256, 0, stream>>>(Wv, wvT, D_HEAD, DMODEL);
  cvt_transpose_kernel<<<256, 256, 0, stream>>>(Wu, wuT, DMODEL, D_HEAD);

  qkv_proj_kernel<<<dim3(NROWS / 16, NHEAD, 3), 256, 0, stream>>>(
      xh, wqT, wkT, wvT, qf, kf, vfT);

  flash_attn_kernel<<<dim3(SEQ / 128, BATCH * NHEAD), 256, 0, stream>>>(
      qf, kf, vfT, attn);

  out_proj_kernel<<<dim3(NROWS / 16), 256, 0, stream>>>(attn, wuT, bu, out);
}
